// ParticleConservingFlow_81423989998120
// MI455X (gfx1250) — compile-verified, matrix operand order checked by reading, not run on hardware
//
#include <hip/hip_runtime.h>
#include <hip/hip_bf16.h>
#include <math.h>

typedef __attribute__((ext_vector_type(16))) __bf16 v16bf;
typedef __attribute__((ext_vector_type(8)))  __bf16 v8bf;
typedef __attribute__((ext_vector_type(8)))  float  v8f;

#define WAVE_LDS_FENCE() asm volatile("s_wait_dscnt 0x0" ::: "memory")

// ---------- LDS layout ----------
// bf16 weights (pre-permuted into WMMA B-fragment order):
//   W1  64x128 : 8192 el | W2 128x64 : 8192 | V1b 64x256 : 16384
//   V2 256x256 : 65536   | V3 256x64 : 16384
// f32: b1(128) b2(64) c1(256) c2(256) c3(64) aLog(64) = 832 floats
// per-wave stage: 5120 bf16 (10240 B) x 8 waves
static constexpr int    OFF_W1 = 0;
static constexpr int    OFF_W2 = 8192;
static constexpr int    OFF_V1 = 16384;
static constexpr int    OFF_V2 = 32768;
static constexpr int    OFF_V3 = 98304;
static constexpr int    NBF16W = 114688;
static constexpr size_t SMEM_BYTES = (size_t)NBF16W * 2 + 832 * 4 + 8 * 10240; // 314624

__device__ __forceinline__ v8f wmma_bf16(v16bf a, v16bf b, v8f c) {
  return __builtin_amdgcn_wmma_f32_16x16x32_bf16(false, a, false, b, (short)0, c,
                                                 false, false);
}

// Fast transcendentals: single v_exp_f32 / v_log_f32 / v_rcp_f32.
__device__ __forceinline__ float fexp(float x) { return __expf(x); }
__device__ __forceinline__ float flog(float x) { return __logf(x); }
__device__ __forceinline__ float frcp(float x) { return __builtin_amdgcn_rcpf(x); }
__device__ __forceinline__ float silu(float x) { return x * frcp(1.0f + fexp(-x)); }

// A fragment from a row-major bf16 tile (16 x K), two 16B LDS loads.
__device__ __forceinline__ v16bf load_a_frag(const __bf16* t, int ld, int k0, int lane) {
  const __bf16* p = t + (lane & 15) * ld + k0 + ((lane >> 4) << 3);
  v8bf lo = *(const v8bf*)p;
  v8bf hi = *(const v8bf*)(p + 16);
  return __builtin_shufflevector(lo, hi, 0, 1, 2, 3, 4, 5, 6, 7, 8, 9, 10, 11, 12, 13, 14, 15);
}

// B fragment from a pre-permuted weight matrix: block (kt,nt) of 512 bf16,
// lane-contiguous; two 16B LDS loads.
__device__ __forceinline__ v16bf load_b_perm(const __bf16* w, int ntiles, int kt, int nt,
                                             int lane) {
  const __bf16* p = w + (((kt * ntiles + nt) << 9) + (lane << 4));
  v8bf lo = *(const v8bf*)p;
  v8bf hi = *(const v8bf*)(p + 8);
  return __builtin_shufflevector(lo, hi, 0, 1, 2, 3, 4, 5, 6, 7, 8, 9, 10, 11, 12, 13, 14, 15);
}

// Cooperative f32 -> bf16 weight stage with B-fragment permutation.
// ISA 16-bit B layout: lane l holds col n=l&15, kb=(l>>4)*8; frag elem j maps to
// K = kt*32 + kb + (j<8 ? j : j+8).
__device__ __forceinline__ void stage_weight_perm(__bf16* dst, const float* src, int K,
                                                  int N, int tid) {
  const int total = K * N;
  const int ntiles = N >> 4;
  for (int d = tid; d < total; d += 256) {
    int blk = d >> 9;
    int within = d & 511;
    int l = within >> 4;
    int j = within & 15;
    int kt = blk / ntiles;
    int nt = blk - kt * ntiles;
    int srcRow = kt * 32 + ((l >> 4) << 3) + (j < 8 ? j : j + 8);
    int srcCol = (nt << 4) + (l & 15);
    dst[d] = (__bf16)src[srcRow * N + srcCol];
  }
}

__device__ __forceinline__ float wred_sum(float v) {
#pragma unroll
  for (int off = 16; off >= 1; off >>= 1) v += __shfl_xor(v, off, 32);
  return v;
}
__device__ __forceinline__ float wred_max(float v) {
#pragma unroll
  for (int off = 16; off >= 1; off >>= 1) v = fmaxf(v, __shfl_xor(v, off, 32));
  return v;
}

// One row of: gumbel top-5 (straight-through config values) + exact PL log-prob
// over 120 permutations. lrow: 64 logits in LDS. Lane handles cols lane, lane+32.
__device__ void gumbel_pl_row(const float* lrow, float u0, float u1, int lane,
                              float* cfg0, float* cfg1, float* lp_out) {
  float l0 = lrow[lane], l1 = lrow[lane + 32];
  float g0 = -flog(-flog(fmaxf(u0, 1e-10f)));
  float g1 = -flog(-flog(fmaxf(u1, 1e-10f)));
  float p0 = l0 + g0, p1 = l1 + g1;  // TEMP = 1
  float E0 = fexp(p0), E1 = fexp(p1);
  float SE = wred_sum(E0 + E1);               // softmax denom (perturbed)
  float T = wred_sum(fexp(l0) + fexp(l1));    // sum exp(plain logits)

  bool a0 = true, a1 = true, s0 = false, s1 = false;
  float S = 0.f;
  float e[5];
#pragma unroll
  for (int k = 0; k < 5; k++) {
    float bv = -3.4e38f;
    int bc = 1 << 30;
    if (a0) { bv = p0; bc = lane; }
    if (a1 && (p1 > bv)) { bv = p1; bc = lane + 32; }
#pragma unroll
    for (int off = 16; off >= 1; off >>= 1) {
      float ov = __shfl_xor(bv, off, 32);
      int oc = __shfl_xor(bc, off, 32);
      if (ov > bv || (ov == bv && oc < bc)) { bv = ov; bc = oc; }
    }
    if (bc == lane) { s0 = true; a0 = false; }
    if (bc == lane + 32) { s1 = true; a1 = false; }
    float lv = lrow[bc];  // wave-uniform LDS broadcast
    S += lv;
    e[k] = fexp(lv);
  }

  float rSE = frcp(SE);
  float sm0 = E0 * rSE, sm1 = E1 * rSE;
  *cfg0 = s0 ? ((1.0f - sm0) + sm0) : 0.0f;  // straight-through forward value
  *cfg1 = s1 ? ((1.0f - sm1) + sm1) : 0.0f;

  // lp(perm) = S - log T - sum_{t=1..4} log(T - prefix_t); logsumexp over 120 perms.
  float base = S - flog(T);
  float f[4];
#pragma unroll
  for (int it = 0; it < 4; it++) {
    int p = lane + 32 * it;
    if (p < 120) {
      int d0 = p / 24, r = p % 24;
      int d1 = r / 6; r %= 6;
      int d2 = r / 2, d3 = r % 2;
      int dd[4] = {d0, d1, d2, d3};
      float a[5] = {e[0], e[1], e[2], e[3], e[4]};
      float pre = 0.f, acc = 0.f;
#pragma unroll
      for (int t = 0; t < 4; t++) {
        int j = dd[t];
        pre += a[j];
        for (int q = j; q < 4; q++) a[q] = a[q + 1];
        acc -= flog(T - pre);
      }
      f[it] = acc;
    } else {
      f[it] = -3.4e38f;
    }
  }
  float mx = wred_max(fmaxf(fmaxf(f[0], f[1]), fmaxf(f[2], f[3])));
  float sum = 0.f;
#pragma unroll
  for (int it = 0; it < 4; it++) sum += fexp(f[it] - mx);
  sum = wred_sum(sum);
  *lp_out = base + mx + flog(sum);
}

__global__ __launch_bounds__(256, 1) void pcf_fused_kernel(
    const float* __restrict__ uA, const float* __restrict__ uB,
    const float* __restrict__ aLog, const float* __restrict__ W1,
    const float* __restrict__ b1, const float* __restrict__ W2,
    const float* __restrict__ b2, const float* __restrict__ V1,
    const float* __restrict__ c1, const float* __restrict__ V2,
    const float* __restrict__ c2, const float* __restrict__ V3,
    const float* __restrict__ c3, float* __restrict__ out, int Brows) {
  extern __shared__ __align__(16) unsigned char smem[];
  __bf16* sW1 = (__bf16*)smem + OFF_W1;
  __bf16* sW2 = (__bf16*)smem + OFF_W2;
  __bf16* sV1 = (__bf16*)smem + OFF_V1;
  __bf16* sV2 = (__bf16*)smem + OFF_V2;
  __bf16* sV3 = (__bf16*)smem + OFF_V3;
  float* fb = (float*)((__bf16*)smem + NBF16W);
  float* sb1 = fb;         // 128
  float* sb2 = fb + 128;   // 64
  float* sc1 = fb + 192;   // 256
  float* sc2 = fb + 448;   // 256
  float* sc3 = fb + 704;   // 64
  float* sAL = fb + 768;   // 64
  __bf16* stage0 = (__bf16*)(fb + 832);

  const int tid = threadIdx.x;
  stage_weight_perm(sW1, W1, 64, 128, tid);
  stage_weight_perm(sW2, W2, 128, 64, tid);
  stage_weight_perm(sV1, V1 + 64 * 256, 64, 256, tid);  // only non-zero rows of V1
  stage_weight_perm(sV2, V2, 256, 256, tid);
  stage_weight_perm(sV3, V3, 256, 64, tid);
  for (int i = tid; i < 128; i += 256) sb1[i] = b1[i];
  for (int i = tid; i < 64; i += 256) sb2[i] = b2[i];
  for (int i = tid; i < 256; i += 256) sc1[i] = c1[i];
  for (int i = tid; i < 256; i += 256) sc2[i] = c2[i];
  for (int i = tid; i < 64; i += 256) sc3[i] = c3[i];
  for (int i = tid; i < 64; i += 256) sAL[i] = aLog[i];
  __syncthreads();

  const int wave = tid >> 5, lane = tid & 31;
  const int n = lane & 15, mb = (lane >> 4) << 3;
  __bf16* stg = stage0 + wave * 5120;  // 10KB per wave
  float* stgF = (float*)stg;           // f32 view for beta logits
  __bf16* stgH = stg + 1024;           // h (16x128)
  __bf16* stgC = stg + 4096;           // alpha_context (16x64)
  const int rowBase = blockIdx.x * 128 + wave * 16;
  if (rowBase >= Brows) return;
  float* outCfg = out;
  float* outLp = out + (size_t)Brows * 128;

  // ---- alpha gumbel top-k + PL, write configs[:,0:64] + bf16 A stage ----
  float lpa_reg = 0.f;
  for (int m = 0; m < 16; m++) {
    int row = rowBase + m;
    float u0 = uA[row * 64 + lane], u1 = uA[row * 64 + lane + 32];
    float c0v, c1v, lp;
    gumbel_pl_row(sAL, u0, u1, lane, &c0v, &c1v, &lp);
    outCfg[(size_t)row * 128 + lane] = c0v;
    outCfg[(size_t)row * 128 + lane + 32] = c1v;
    stg[m * 64 + lane] = (__bf16)c0v;
    stg[m * 64 + lane + 32] = (__bf16)c1v;
    if (lane == m) lpa_reg = lp;
  }
  WAVE_LDS_FENCE();

  // ---- GEMM1: h = silu(cfg(16x64) @ W1(64x128) + b1) ----
  {
    v16bf a0 = load_a_frag(stg, 64, 0, lane);
    v16bf a1 = load_a_frag(stg, 64, 32, lane);
#pragma unroll
    for (int nt = 0; nt < 8; nt++) {
      v8f acc = {0.f, 0.f, 0.f, 0.f, 0.f, 0.f, 0.f, 0.f};
      acc = wmma_bf16(a0, load_b_perm(sW1, 8, 0, nt, lane), acc);
      acc = wmma_bf16(a1, load_b_perm(sW1, 8, 1, nt, lane), acc);
      float bias = sb1[nt * 16 + n];
#pragma unroll
      for (int j = 0; j < 8; j++)
        stgH[(mb + j) * 128 + nt * 16 + n] = (__bf16)silu(acc[j] + bias);
    }
  }
  WAVE_LDS_FENCE();

  // ---- GEMM2: ctx = h(16x128) @ W2(128x64) + b2 ----
  {
    v16bf ha[4];
#pragma unroll
    for (int kt = 0; kt < 4; kt++) ha[kt] = load_a_frag(stgH, 128, kt * 32, lane);
#pragma unroll
    for (int nt = 0; nt < 4; nt++) {
      v8f acc = {0.f, 0.f, 0.f, 0.f, 0.f, 0.f, 0.f, 0.f};
#pragma unroll
      for (int kt = 0; kt < 4; kt++)
        acc = wmma_bf16(ha[kt], load_b_perm(sW2, 4, kt, nt, lane), acc);
      float bias = sb2[nt * 16 + n];
#pragma unroll
      for (int j = 0; j < 8; j++)
        stgC[(mb + j) * 64 + nt * 16 + n] = (__bf16)(acc[j] + bias);
    }
  }
  WAVE_LDS_FENCE();

  // ---- GEMM3: z1 = silu(ctx(16x64) @ V1b(64x256) + c1), overwrites cfg/h ----
  {
    v16bf ca0 = load_a_frag(stgC, 64, 0, lane);
    v16bf ca1 = load_a_frag(stgC, 64, 32, lane);
    WAVE_LDS_FENCE();
#pragma unroll
    for (int nt = 0; nt < 16; nt++) {
      v8f acc = {0.f, 0.f, 0.f, 0.f, 0.f, 0.f, 0.f, 0.f};
      acc = wmma_bf16(ca0, load_b_perm(sV1, 16, 0, nt, lane), acc);
      acc = wmma_bf16(ca1, load_b_perm(sV1, 16, 1, nt, lane), acc);
      float bias = sc1[nt * 16 + n];
#pragma unroll
      for (int j = 0; j < 8; j++)
        stg[(mb + j) * 256 + nt * 16 + n] = (__bf16)silu(acc[j] + bias);
    }
  }
  WAVE_LDS_FENCE();

  // ---- GEMM4: z2 = silu(z1(16x256) @ V2(256x256) + c2); acc in regs, in-place ----
  {
    v8f acc[16];
#pragma unroll
    for (int nt = 0; nt < 16; nt++) acc[nt] = v8f{0.f, 0.f, 0.f, 0.f, 0.f, 0.f, 0.f, 0.f};
#pragma unroll
    for (int kt = 0; kt < 8; kt++) {
      v16bf a = load_a_frag(stg, 256, kt * 32, lane);
#pragma unroll
      for (int nt = 0; nt < 16; nt++)
        acc[nt] = wmma_bf16(a, load_b_perm(sV2, 16, kt, nt, lane), acc[nt]);
    }
    WAVE_LDS_FENCE();  // all z1 reads done before overwrite
#pragma unroll
    for (int nt = 0; nt < 16; nt++) {
      float bias = sc2[nt * 16 + n];
#pragma unroll
      for (int j = 0; j < 8; j++)
        stg[(mb + j) * 256 + nt * 16 + n] = (__bf16)silu(acc[nt][j] + bias);
    }
  }
  WAVE_LDS_FENCE();

  // ---- GEMM5: beta_logits = z2(16x256) @ V3(256x64) + c3 -> f32 stage ----
  {
    v8f acc[4];
#pragma unroll
    for (int nt = 0; nt < 4; nt++) acc[nt] = v8f{0.f, 0.f, 0.f, 0.f, 0.f, 0.f, 0.f, 0.f};
#pragma unroll
    for (int kt = 0; kt < 8; kt++) {
      v16bf a = load_a_frag(stg, 256, kt * 32, lane);
#pragma unroll
      for (int nt = 0; nt < 4; nt++)
        acc[nt] = wmma_bf16(a, load_b_perm(sV3, 4, kt, nt, lane), acc[nt]);
    }
    WAVE_LDS_FENCE();
#pragma unroll
    for (int nt = 0; nt < 4; nt++) {
      float bias = sc3[nt * 16 + n];
#pragma unroll
      for (int j = 0; j < 8; j++)
        stgF[(mb + j) * 64 + nt * 16 + n] = acc[nt][j] + bias;
    }
  }
  WAVE_LDS_FENCE();

  // ---- beta gumbel top-k + PL, write configs[:,64:128] + log_probs ----
  for (int m = 0; m < 16; m++) {
    int row = rowBase + m;
    float u0 = uB[row * 64 + lane], u1 = uB[row * 64 + lane + 32];
    float c0v, c1v, lp;
    gumbel_pl_row(stgF + m * 64, u0, u1, lane, &c0v, &c1v, &lp);
    outCfg[(size_t)row * 128 + 64 + lane] = c0v;
    outCfg[(size_t)row * 128 + 96 + lane] = c1v;
    float lpa = __shfl(lpa_reg, m, 32);
    if (lane == 0) outLp[row] = lpa + lp;
  }
}

extern "C" void kernel_launch(void* const* d_in, const int* in_sizes, int n_in,
                              void* d_out, int out_size, void* d_ws, size_t ws_size,
                              hipStream_t stream) {
  const float* uA = (const float*)d_in[0];
  const float* uB = (const float*)d_in[1];
  const float* aL = (const float*)d_in[2];
  const float* W1 = (const float*)d_in[3];
  const float* b1 = (const float*)d_in[4];
  const float* W2 = (const float*)d_in[5];
  const float* b2 = (const float*)d_in[6];
  const float* V1 = (const float*)d_in[7];
  const float* c1 = (const float*)d_in[8];
  const float* V2 = (const float*)d_in[9];
  const float* c2 = (const float*)d_in[10];
  const float* V3 = (const float*)d_in[11];
  const float* c3 = (const float*)d_in[12];
  float* out = (float*)d_out;

  const int Brows = in_sizes[0] / 64;          // 8192
  const int grid = (Brows + 127) / 128;        // 128 rows per 8-wave block
  pcf_fused_kernel<<<grid, 256, SMEM_BYTES, stream>>>(
      uA, uB, aL, W1, b1, W2, b2, V1, c1, V2, c2, V3, c3, out, Brows);
}
